// UpdateFunction_28028956573797
// MI455X (gfx1250) — matrix operand order repaired and checked by reading, not compiled, not run on hardware
//
#include <hip/hip_runtime.h>

typedef __attribute__((ext_vector_type(8)))  float   v8f;
typedef __attribute__((ext_vector_type(4)))  float   f4;
typedef __attribute__((ext_vector_type(16))) __bf16  bf16x16;
typedef __attribute__((ext_vector_type(8)))  __bf16  bf16x8;
typedef __attribute__((ext_vector_type(4)))  __bf16  bf16x4;

#define DD   256      // node dim
#define KM   512      // message dim (2D)
#define KX   768      // concatenated K = 2D + D
#define MT   64       // rows per workgroup
#define XSTR 776      // LDS row stride for X (bf16 elems), +8 pad -> 4-bank rotation
#define RSTR 264      // LDS row stride for RH

union Frag { bf16x16 v; bf16x8 q[2]; };

__device__ __forceinline__ float fsigmoid(float x) {
  return 1.0f / (1.0f + __expf(-x));
}
__device__ __forceinline__ float ftanh_fast(float x) {
  float e = __expf(2.0f * x);
  return 1.0f - 2.0f / (e + 1.0f);
}

// ---------------------------------------------------------------------------
// Prep: build Wt[3][256][768] bf16, N-major:  Wt[mat][n][k]
//   k < 512 -> W[k][n] ; k >= 512 -> U[k-512][n]
//   mat: 0 = (w_z,u_z), 1 = (w_r,u_r), 2 = (w,u)
// ---------------------------------------------------------------------------
__global__ __launch_bounds__(256) void gru_prep(
    const float* __restrict__ wz, const float* __restrict__ uz,
    const float* __restrict__ wr, const float* __restrict__ ur,
    const float* __restrict__ w,  const float* __restrict__ u,
    __bf16* __restrict__ Wt)
{
  int idx = blockIdx.x * 256 + threadIdx.x;
  if (idx >= 3 * DD * KX) return;
  int k   = idx % KX;
  int n   = (idx / KX) % DD;
  int mat = idx / (KX * DD);
  const float* W = (mat == 0) ? wz : (mat == 1) ? wr : w;
  const float* U = (mat == 0) ? uz : (mat == 1) ? ur : u;
  float v = (k < KM) ? W[k * DD + n] : U[(k - KM) * DD + n];
  Wt[idx] = (__bf16)v;
}

// ---------------------------------------------------------------------------
// Main fused GRU update kernel
// ---------------------------------------------------------------------------
__global__ __launch_bounds__(256) void gru_main(
    const float* __restrict__ hglob,
    const float* __restrict__ mglob,
    const float* __restrict__ mask,
    const __bf16* __restrict__ Wt,
    float* __restrict__ out)
{
  __shared__ __bf16 X[MT][XSTR];    // [m | h] tile, bf16
  __shared__ __bf16 RH[MT][RSTR];   // r_t * h tile, bf16
  __shared__ float  maskS[MT];

  const int tid  = threadIdx.x;
  const int wave = tid >> 5;
  const int lane = tid & 31;
  const int l15  = lane & 15;
  const int lhi  = lane >> 4;
  const long rowBase = (long)blockIdx.x * MT;

  // ---- Stage X = [m | h] into LDS as bf16 -------------------------------
  {
    const f4* m4 = (const f4*)(mglob + rowBase * KM);
    #pragma unroll
    for (int it = 0; it < 32; ++it) {           // 64*512 floats / 4 / 256
      int idx = tid + it * 256;
      int r = idx >> 7, c = idx & 127;          // c in f4 units (128 per row)
      f4 v = m4[r * 128 + c];
      bf16x4 b;
      b.x = (__bf16)v.x; b.y = (__bf16)v.y; b.z = (__bf16)v.z; b.w = (__bf16)v.w;
      *(bf16x4*)&X[r][c * 4] = b;
    }
    const f4* h4 = (const f4*)(hglob + rowBase * DD);
    #pragma unroll
    for (int it = 0; it < 16; ++it) {           // 64*256 floats / 4 / 256
      int idx = tid + it * 256;
      int r = idx >> 6, c = idx & 63;
      f4 v = h4[r * 64 + c];
      bf16x4 b;
      b.x = (__bf16)v.x; b.y = (__bf16)v.y; b.z = (__bf16)v.z; b.w = (__bf16)v.w;
      *(bf16x4*)&X[r][KM + c * 4] = b;
    }
    if (tid < MT) maskS[tid] = mask[rowBase + tid];
  }
  __syncthreads();

  // Wave owns columns [nb0, nb0+32): two 16-wide N tiles, all 4 M tiles.
  const int nb0 = wave * 32;
  const __bf16* WtZ = Wt;
  const __bf16* WtR = Wt + 1 * DD * KX;
  const __bf16* WtC = Wt + 2 * DD * KX;

  v8f accZ[4][2], accR[4][2];
  #pragma unroll
  for (int mi = 0; mi < 4; ++mi)
    #pragma unroll
    for (int t = 0; t < 2; ++t) {
      accZ[mi][t] = (v8f){0.f,0.f,0.f,0.f,0.f,0.f,0.f,0.f};
      accR[mi][t] = (v8f){0.f,0.f,0.f,0.f,0.f,0.f,0.f,0.f};
    }

  // ---- Pass 1+2 fused: z and r GEMMs over K = 768 ------------------------
  for (int ks = 0; ks < KX / 32; ++ks) {
    const int k0 = ks * 32;
    Frag a[4];
    #pragma unroll
    for (int mi = 0; mi < 4; ++mi) {
      const __bf16* p = &X[mi * 16 + l15][k0 + lhi * 8];
      a[mi].q[0] = *(const bf16x8*)p;           // K + {0..7 | 8..15}
      a[mi].q[1] = *(const bf16x8*)(p + 16);    // K + {16..23 | 24..31}
    }
    #pragma unroll
    for (int t = 0; t < 2; ++t) {
      const int n = nb0 + t * 16 + l15;
      Frag bz, br;
      const __bf16* pz = WtZ + n * KX + k0 + lhi * 8;  // B: K + {0..7 | 8..15}, {16..23 | 24..31}
      bz.q[0] = *(const bf16x8*)pz;
      bz.q[1] = *(const bf16x8*)(pz + 16);
      const __bf16* pr = WtR + n * KX + k0 + lhi * 8;
      br.q[0] = *(const bf16x8*)pr;
      br.q[1] = *(const bf16x8*)(pr + 16);
      #pragma unroll
      for (int mi = 0; mi < 4; ++mi) {
        accZ[mi][t] = __builtin_amdgcn_wmma_f32_16x16x32_bf16(
            false, a[mi].v, false, bz.v, (short)0, accZ[mi][t], false, false);
        accR[mi][t] = __builtin_amdgcn_wmma_f32_16x16x32_bf16(
            false, a[mi].v, false, br.v, (short)0, accR[mi][t], false, false);
      }
    }
  }

  // ---- z = sigmoid(.) kept in regs;  RH = sigmoid(r) * h  -> LDS ---------
  #pragma unroll
  for (int mi = 0; mi < 4; ++mi)
    #pragma unroll
    for (int t = 0; t < 2; ++t)
      #pragma unroll
      for (int v = 0; v < 8; ++v) {
        const int row = mi * 16 + v + lhi * 8;
        const int col = nb0 + t * 16 + l15;
        accZ[mi][t][v] = fsigmoid(accZ[mi][t][v]);
        float rs = fsigmoid(accR[mi][t][v]);
        float hv = (float)X[row][KM + col];
        RH[row][col] = (__bf16)(rs * hv);
      }
  __syncthreads();

  // ---- Pass 3: candidate GEMM over [m | r*h] -----------------------------
  v8f accH[4][2];
  #pragma unroll
  for (int mi = 0; mi < 4; ++mi)
    #pragma unroll
    for (int t = 0; t < 2; ++t)
      accH[mi][t] = (v8f){0.f,0.f,0.f,0.f,0.f,0.f,0.f,0.f};

  // K = 0..511 from m-part of X
  for (int ks = 0; ks < KM / 32; ++ks) {
    const int k0 = ks * 32;
    Frag a[4];
    #pragma unroll
    for (int mi = 0; mi < 4; ++mi) {
      const __bf16* p = &X[mi * 16 + l15][k0 + lhi * 8];
      a[mi].q[0] = *(const bf16x8*)p;
      a[mi].q[1] = *(const bf16x8*)(p + 16);
    }
    #pragma unroll
    for (int t = 0; t < 2; ++t) {
      const int n = nb0 + t * 16 + l15;
      Frag bc;
      const __bf16* pc = WtC + n * KX + k0 + lhi * 8;
      bc.q[0] = *(const bf16x8*)pc;
      bc.q[1] = *(const bf16x8*)(pc + 16);
      #pragma unroll
      for (int mi = 0; mi < 4; ++mi)
        accH[mi][t] = __builtin_amdgcn_wmma_f32_16x16x32_bf16(
            false, a[mi].v, false, bc.v, (short)0, accH[mi][t], false, false);
    }
  }
  // K = 512..767 from RH
  for (int ks = 0; ks < DD / 32; ++ks) {
    const int kl = ks * 32;
    const int k0 = KM + kl;
    Frag a[4];
    #pragma unroll
    for (int mi = 0; mi < 4; ++mi) {
      const __bf16* p = &RH[mi * 16 + l15][kl + lhi * 8];
      a[mi].q[0] = *(const bf16x8*)p;
      a[mi].q[1] = *(const bf16x8*)(p + 16);
    }
    #pragma unroll
    for (int t = 0; t < 2; ++t) {
      const int n = nb0 + t * 16 + l15;
      Frag bc;
      const __bf16* pc = WtC + n * KX + k0 + lhi * 8;
      bc.q[0] = *(const bf16x8*)pc;
      bc.q[1] = *(const bf16x8*)(pc + 16);
      #pragma unroll
      for (int mi = 0; mi < 4; ++mi)
        accH[mi][t] = __builtin_amdgcn_wmma_f32_16x16x32_bf16(
            false, a[mi].v, false, bc.v, (short)0, accH[mi][t], false, false);
    }
  }

  // ---- Epilogue: h_t = ((1-z)*h + z*tanh(acc)) * mask --------------------
  #pragma unroll
  for (int mi = 0; mi < 4; ++mi)
    #pragma unroll
    for (int t = 0; t < 2; ++t)
      #pragma unroll
      for (int v = 0; v < 8; ++v) {
        const int row = mi * 16 + v + lhi * 8;
        const int col = nb0 + t * 16 + l15;
        float z  = accZ[mi][t][v];
        float ht = ftanh_fast(accH[mi][t][v]);
        float hv = (float)X[row][KM + col];
        float o  = ((1.0f - z) * hv + z * ht) * maskS[row];
        out[(rowBase + row) * DD + col] = o;
      }
}

// ---------------------------------------------------------------------------
extern "C" void kernel_launch(void* const* d_in, const int* in_sizes, int n_in,
                              void* d_out, int out_size, void* d_ws, size_t ws_size,
                              hipStream_t stream) {
  const float* node_state = (const float*)d_in[0];
  const float* message    = (const float*)d_in[1];
  const float* mask       = (const float*)d_in[2];
  const float* w_z = (const float*)d_in[3];
  const float* u_z = (const float*)d_in[4];
  const float* w_r = (const float*)d_in[5];
  const float* u_r = (const float*)d_in[6];
  const float* w   = (const float*)d_in[7];
  const float* u   = (const float*)d_in[8];
  float*  out = (float*)d_out;
  __bf16* Wt  = (__bf16*)d_ws;          // needs 3*256*768*2 = 1,179,648 bytes

  const int prep_elems  = 3 * DD * KX;
  const int prep_blocks = (prep_elems + 255) / 256;
  gru_prep<<<prep_blocks, 256, 0, stream>>>(w_z, u_z, w_r, u_r, w, u, Wt);

  const int rows = 64 * 1024;           // BATCH * NUM_NODES
  gru_main<<<rows / MT, 256, 0, stream>>>(node_state, message, mask, Wt, out);
}
